// MultiHeadAttention_10694468567717
// MI455X (gfx1250) — matrix operand order repaired and checked by reading, not compiled, not run on hardware
//
#include <hip/hip_runtime.h>
#include <hip/hip_fp16.h>

#define D_MODEL 512
#define N_HEADS 8
#define D_HEAD  64
#define BATCH   2
#define SEQ     4096
#define BS      (BATCH*SEQ)

typedef __attribute__((ext_vector_type(16))) _Float16 v16h;
typedef __attribute__((ext_vector_type(8)))  _Float16 v8h;
typedef __attribute__((ext_vector_type(8)))  float    v8f;
typedef __attribute__((ext_vector_type(4)))  float    v4f;

static __device__ __forceinline__ v8h cvt8_f32_f16(const float* p) {
  v4f a = *(const v4f*)p;
  v4f b = *(const v4f*)(p + 4);
  v8h r;
  r[0]=(_Float16)a[0]; r[1]=(_Float16)a[1]; r[2]=(_Float16)a[2]; r[3]=(_Float16)a[3];
  r[4]=(_Float16)b[0]; r[5]=(_Float16)b[1]; r[6]=(_Float16)b[2]; r[7]=(_Float16)b[3];
  return r;
}

static __device__ __forceinline__ v16h join16(v8h lo, v8h hi) {
  v16h r;
#pragma unroll
  for (int i = 0; i < 8; i++) { r[i] = lo[i]; r[8+i] = hi[i]; }
  return r;
}

// async 16B global -> LDS copy (per-lane addresses), tracked by ASYNCcnt
static __device__ __forceinline__ void async_copy_b128(unsigned lds_addr,
                                                       const void* gptr) {
  unsigned long long ga = (unsigned long long)(uintptr_t)gptr;
  asm volatile("global_load_async_to_lds_b128 %0, %1, off"
               :: "v"(lds_addr), "v"(ga) : "memory");
}
static __device__ __forceinline__ void wait_asynccnt0() {
  asm volatile("s_wait_asynccnt 0x0" ::: "memory");
}

// ---------------------------------------------------------------------------
// Kernel 1: fused QKV projection.  y = x @ W^T + b  (f32 in, f16 out)
// grid (BS/16, D_MODEL/128, 3), block 256 (8 waves); wave -> one 16x16 tile.
// Output layout: [b][h][s][dh]  (f16)
// ---------------------------------------------------------------------------
__global__ __launch_bounds__(256)
void qkv_proj_kernel(const float* __restrict__ x,
                     const float* __restrict__ Wq, const float* __restrict__ Wk,
                     const float* __restrict__ Wv,
                     const float* __restrict__ bq, const float* __restrict__ bk,
                     const float* __restrict__ bv,
                     _Float16* __restrict__ Qh, _Float16* __restrict__ Kh,
                     _Float16* __restrict__ Vh)
{
  const float* W; const float* bias; _Float16* out;
  if      (blockIdx.z == 0) { W = Wq; bias = bq; out = Qh; }
  else if (blockIdx.z == 1) { W = Wk; bias = bk; out = Kh; }
  else                      { W = Wv; bias = bv; out = Vh; }

  __shared__ _Float16 sB[32][136];   // transposed W tile: [e(=K)][f], padded

  const int tid  = threadIdx.x;
  const int wave = tid >> 5;
  const int lane = tid & 31;
  const int l16  = lane & 15;
  const int h2   = lane >> 4;

  const int rowBase = blockIdx.x * 16;
  const int colBlk  = blockIdx.y * 128;
  const int colBase = colBlk + wave * 16;
  const int mrow    = rowBase + l16;

  v8f acc = {};

  for (int k = 0; k < D_MODEL; k += 32) {
    // cooperative: W[colBlk+f][k+e] -> sB[e][f]  (f 0..127, e 0..31)
    {
      const int f    = tid >> 1;
      const int eseg = (tid & 1) * 16;
      const float* wp = W + (size_t)(colBlk + f) * D_MODEL + k + eseg;
      v8h w0 = cvt8_f32_f16(wp);
      v8h w1 = cvt8_f32_f16(wp + 8);
#pragma unroll
      for (int i = 0; i < 8; i++) sB[eseg + i][f]     = w0[i];
#pragma unroll
      for (int i = 0; i < 8; i++) sB[eseg + 8 + i][f] = w1[i];
    }
    __syncthreads();

    // A fragment from x: lane row = mrow, K runs at k+h2*8 and k+16+h2*8
    const float* ap = x + (size_t)mrow * D_MODEL + k + h2 * 8;
    v16h afrag = join16(cvt8_f32_f16(ap), cvt8_f32_f16(ap + 16));

    // B fragment from LDS: lane = output col, elements = K (table order)
    const _Float16* bp = &sB[h2 * 8][wave * 16 + l16];
    v16h bfrag;
#pragma unroll
    for (int i = 0; i < 8; i++) { bfrag[i] = bp[i * 136]; bfrag[8 + i] = bp[(16 + i) * 136]; }

    acc = __builtin_amdgcn_wmma_f32_16x16x32_f16(false, afrag, false, bfrag,
                                                 (short)0, acc, false, false);
    __syncthreads();
  }

  const int ncol = colBase + l16;
  const float bb_ = bias[ncol];
  const int hh = ncol / D_HEAD;
  const int dd = ncol % D_HEAD;
  const int bb = rowBase / SEQ;   // 16-row tile never crosses a batch boundary
#pragma unroll
  for (int r = 0; r < 8; r++) {
    int m = rowBase + r + 8 * h2;
    int s = m - bb * SEQ;
    out[(((size_t)bb * N_HEADS + hh) * SEQ + s) * D_HEAD + dd] =
        (_Float16)(acc[r] + bb_);
  }
}

// ---------------------------------------------------------------------------
// Kernel 2: flash attention.  grid (SEQ/64, H, B), block 128 (4 waves).
// Each wave: 16 query rows, full D_HEAD=64, online softmax over 32-key steps.
// V tile is staged with GLOBAL_LOAD_ASYNC_TO_LDS_B128 (ASYNCcnt path).
// ---------------------------------------------------------------------------
__global__ __launch_bounds__(128)
void flash_attn_kernel(const _Float16* __restrict__ Qh,
                       const _Float16* __restrict__ Kh,
                       const _Float16* __restrict__ Vh,
                       const float*    __restrict__ mask,
                       _Float16*       __restrict__ Oh)
{
  __shared__ _Float16 sKT[64][40];      // transposed K tile [d][key], padded
  __shared__ _Float16 sV [32][72];      // V tile [key][d], padded
  __shared__ _Float16 sP [4][16][32];   // per-wave probability tile

  const int tid  = threadIdx.x;
  const int wave = tid >> 5;
  const int lane = tid & 31;
  const int l16  = lane & 15;
  const int h2   = lane >> 4;

  const int h     = blockIdx.y;
  const int b     = blockIdx.z;
  const int qBase = blockIdx.x * 64 + wave * 16;
  const size_t headOff = ((size_t)b * N_HEADS + h) * SEQ;

  // persistent Q fragments (A layout), d 0..31 and 32..63
  const _Float16* qp = Qh + (headOff + qBase + l16) * D_HEAD + h2 * 8;
  v16h qa0 = join16(*(const v8h*)qp,        *(const v8h*)(qp + 16));
  v16h qa1 = join16(*(const v8h*)(qp + 32), *(const v8h*)(qp + 48));

  float mrow[8], lrow[8];
#pragma unroll
  for (int r = 0; r < 8; r++) { mrow[r] = -1e30f; lrow[r] = 0.0f; }
  v8f o0 = {}, o1 = {}, o2 = {}, o3 = {};

  const float scale = 0.125f;  // 1/sqrt(D_HEAD)

  const int ldRow = tid >> 2;         // 0..31
  const int ldSeg = (tid & 3) * 16;   // 0,16,32,48

  const unsigned vDst0 = (unsigned)(uintptr_t)&sV[ldRow][ldSeg];
  const unsigned vDst1 = (unsigned)(uintptr_t)&sV[ldRow][ldSeg + 8];

  for (int kv = 0; kv < SEQ; kv += 32) {
    __syncthreads();
    {
      // V tile: pure byte copy, natural layout -> async global->LDS (B128 x2)
      const _Float16* vp = Vh + (headOff + kv + ldRow) * D_HEAD + ldSeg;
      async_copy_b128(vDst0, vp);
      async_copy_b128(vDst1, vp + 8);

      // K tile: needs transpose for the score B-layout -> VALU/DS path
      const _Float16* kp = Kh + (headOff + kv + ldRow) * D_HEAD + ldSeg;
      v8h k0 = *(const v8h*)kp;
      v8h k1 = *(const v8h*)(kp + 8);
#pragma unroll
      for (int i = 0; i < 8; i++) sKT[ldSeg + i][ldRow]     = k0[i];
#pragma unroll
      for (int i = 0; i < 8; i++) sKT[ldSeg + 8 + i][ldRow] = k1[i];

      if (kv + 32 < SEQ) {                     // global_prefetch_b8 next tiles
        __builtin_prefetch(kp + 32 * D_HEAD, 0, 1);
        __builtin_prefetch(vp + 32 * D_HEAD, 0, 1);
      }
    }
    wait_asynccnt0();     // V bytes resident in LDS before the barrier
    __syncthreads();

    // scores: two 16x16 tiles (keys kv+0..15 and kv+16..31)
    v8f s0 = {}, s1 = {};
    {
      const _Float16* ktA = &sKT[h2 * 8][l16];       // key = l16, d table order (vs qa0)
      const _Float16* ktB = &sKT[32 + h2 * 8][l16];  // key = l16, d = 32 + table order (vs qa1)
      v16h b0c0, b1c0, b0c1, b1c1;
#pragma unroll
      for (int i = 0; i < 8; i++) {
        b0c0[i] = ktA[i * 40];      b0c0[8 + i] = ktA[(16 + i) * 40];
        b1c0[i] = ktB[i * 40];      b1c0[8 + i] = ktB[(16 + i) * 40];
        b0c1[i] = ktA[i * 40 + 16]; b0c1[8 + i] = ktA[(16 + i) * 40 + 16];
        b1c1[i] = ktB[i * 40 + 16]; b1c1[8 + i] = ktB[(16 + i) * 40 + 16];
      }
      s0 = __builtin_amdgcn_wmma_f32_16x16x32_f16(false, qa0, false, b0c0, (short)0, s0, false, false);
      s0 = __builtin_amdgcn_wmma_f32_16x16x32_f16(false, qa1, false, b1c0, (short)0, s0, false, false);
      s1 = __builtin_amdgcn_wmma_f32_16x16x32_f16(false, qa0, false, b0c1, (short)0, s1, false, false);
      s1 = __builtin_amdgcn_wmma_f32_16x16x32_f16(false, qa1, false, b1c1, (short)0, s1, false, false);
    }

    // scale + mask + online softmax
    float tmax[8];
#pragma unroll
    for (int r = 0; r < 8; r++) {
      const float* mp = mask + (size_t)(qBase + r + 8 * h2) * SEQ + kv;
      float v0 = s0[r] * scale + mp[l16];
      float v1 = s1[r] * scale + mp[16 + l16];
      s0[r] = v0; s1[r] = v1;
      tmax[r] = fmaxf(v0, v1);
    }
#pragma unroll
    for (int off = 1; off < 16; off <<= 1)
#pragma unroll
      for (int r = 0; r < 8; r++)
        tmax[r] = fmaxf(tmax[r], __shfl_xor(tmax[r], off, 32));

    float alpha[8], psum[8];
#pragma unroll
    for (int r = 0; r < 8; r++) {
      float mn = fmaxf(mrow[r], tmax[r]);
      alpha[r] = __expf(mrow[r] - mn);
      mrow[r] = mn;
      float p0 = __expf(s0[r] - mn);
      float p1 = __expf(s1[r] - mn);
      sP[wave][r + 8 * h2][l16]      = (_Float16)p0;
      sP[wave][r + 8 * h2][16 + l16] = (_Float16)p1;
      psum[r] = p0 + p1;
    }
#pragma unroll
    for (int off = 1; off < 16; off <<= 1)
#pragma unroll
      for (int r = 0; r < 8; r++)
        psum[r] += __shfl_xor(psum[r], off, 32);
#pragma unroll
    for (int r = 0; r < 8; r++) {
      lrow[r] = lrow[r] * alpha[r] + psum[r];
      o0[r] *= alpha[r]; o1[r] *= alpha[r]; o2[r] *= alpha[r]; o3[r] *= alpha[r];
    }

    // P fragment (A layout) — same-wave LDS traffic, DS ops are in order
    const _Float16* pp = &sP[wave][l16][h2 * 8];
    v16h pfrag = join16(*(const v8h*)pp, *(const v8h*)(pp + 16));

    // V fragments: lane = d col, elements = key (table order)
    const _Float16* vq = &sV[h2 * 8][l16];
    v16h vb0, vb1, vb2, vb3;
#pragma unroll
    for (int i = 0; i < 8; i++) {
      vb0[i] = vq[i * 72];      vb0[8 + i] = vq[(16 + i) * 72];
      vb1[i] = vq[i * 72 + 16]; vb1[8 + i] = vq[(16 + i) * 72 + 16];
      vb2[i] = vq[i * 72 + 32]; vb2[8 + i] = vq[(16 + i) * 72 + 32];
      vb3[i] = vq[i * 72 + 48]; vb3[8 + i] = vq[(16 + i) * 72 + 48];
    }
    o0 = __builtin_amdgcn_wmma_f32_16x16x32_f16(false, pfrag, false, vb0, (short)0, o0, false, false);
    o1 = __builtin_amdgcn_wmma_f32_16x16x32_f16(false, pfrag, false, vb1, (short)0, o1, false, false);
    o2 = __builtin_amdgcn_wmma_f32_16x16x32_f16(false, pfrag, false, vb2, (short)0, o2, false, false);
    o3 = __builtin_amdgcn_wmma_f32_16x16x32_f16(false, pfrag, false, vb3, (short)0, o3, false, false);
  }

  // normalize and store (f16, [b][s][h*64+d] layout for the output GEMM)
#pragma unroll
  for (int r = 0; r < 8; r++) {
    float inv = (lrow[r] > 0.0f) ? (1.0f / lrow[r]) : 0.0f;   // nan_to_num
    int s = qBase + r + 8 * h2;
    size_t base = ((size_t)b * SEQ + s) * D_MODEL + h * D_HEAD;
    Oh[base +  0 + l16] = (_Float16)(o0[r] * inv);
    Oh[base + 16 + l16] = (_Float16)(o1[r] * inv);
    Oh[base + 32 + l16] = (_Float16)(o2[r] * inv);
    Oh[base + 48 + l16] = (_Float16)(o3[r] * inv);
  }
}

// ---------------------------------------------------------------------------
// Kernel 3: output projection.  y = Oh @ Wo^T + bo  (f16 A, f32 out)
// ---------------------------------------------------------------------------
__global__ __launch_bounds__(256)
void out_proj_kernel(const _Float16* __restrict__ Oh,
                     const float* __restrict__ Wo, const float* __restrict__ bo,
                     float* __restrict__ y)
{
  __shared__ _Float16 sB[32][136];

  const int tid  = threadIdx.x;
  const int wave = tid >> 5;
  const int lane = tid & 31;
  const int l16  = lane & 15;
  const int h2   = lane >> 4;

  const int rowBase = blockIdx.x * 16;
  const int colBlk  = blockIdx.y * 128;
  const int colBase = colBlk + wave * 16;
  const int mrow    = rowBase + l16;

  v8f acc = {};

  for (int k = 0; k < D_MODEL; k += 32) {
    {
      const int f    = tid >> 1;
      const int eseg = (tid & 1) * 16;
      const float* wp = Wo + (size_t)(colBlk + f) * D_MODEL + k + eseg;
      v8h w0 = cvt8_f32_f16(wp);
      v8h w1 = cvt8_f32_f16(wp + 8);
#pragma unroll
      for (int i = 0; i < 8; i++) sB[eseg + i][f]     = w0[i];
#pragma unroll
      for (int i = 0; i < 8; i++) sB[eseg + 8 + i][f] = w1[i];
    }
    __syncthreads();

    const _Float16* ap = Oh + (size_t)mrow * D_MODEL + k + h2 * 8;
    v16h afrag = join16(*(const v8h*)ap, *(const v8h*)(ap + 16));

    const _Float16* bp = &sB[h2 * 8][wave * 16 + l16];
    v16h bfrag;
#pragma unroll
    for (int i = 0; i < 8; i++) { bfrag[i] = bp[i * 136]; bfrag[8 + i] = bp[(16 + i) * 136]; }

    acc = __builtin_amdgcn_wmma_f32_16x16x32_f16(false, afrag, false, bfrag,
                                                 (short)0, acc, false, false);
    __syncthreads();
  }

  const int ncol = colBase + l16;
  const float bb_ = bo[ncol];
#pragma unroll
  for (int r = 0; r < 8; r++) {
    int m = rowBase + r + 8 * h2;
    y[(size_t)m * D_MODEL + ncol] = acc[r] + bb_;
  }
}

// ---------------------------------------------------------------------------
extern "C" void kernel_launch(void* const* d_in, const int* in_sizes, int n_in,
                              void* d_out, int out_size, void* d_ws, size_t ws_size,
                              hipStream_t stream)
{
  const float* x    = (const float*)d_in[0];
  const float* mask = (const float*)d_in[1];
  const float* Wq   = (const float*)d_in[2];
  const float* bq   = (const float*)d_in[3];
  const float* Wk   = (const float*)d_in[4];
  const float* bk   = (const float*)d_in[5];
  const float* Wv   = (const float*)d_in[6];
  const float* bv   = (const float*)d_in[7];
  const float* Wo   = (const float*)d_in[8];
  const float* bo   = (const float*)d_in[9];
  float* y = (float*)d_out;

  const size_t headElems = (size_t)BATCH * N_HEADS * SEQ * D_HEAD;   // 4.19M f16
  char* ws = (char*)d_ws;
  _Float16* Qh = (_Float16*)(ws);
  _Float16* Kh = (_Float16*)(ws + 2 * headElems);
  _Float16* Vh = (_Float16*)(ws + 4 * headElems);
  _Float16* Oh = (_Float16*)(ws + 6 * headElems);

  qkv_proj_kernel<<<dim3(BS / 16, D_MODEL / 128, 3), 256, 0, stream>>>(
      x, Wq, Wk, Wv, bq, bk, bv, Qh, Kh, Vh);
  flash_attn_kernel<<<dim3(SEQ / 64, N_HEADS, BATCH), 128, 0, stream>>>(
      Qh, Kh, Vh, mask, Oh);
  out_proj_kernel<<<dim3(BS / 16, D_MODEL / 128), 256, 0, stream>>>(
      Oh, Wo, bo, y);
}
